// SimpleSelfAttention_90065464197561
// MI455X (gfx1250) — compile-verified, matrix operand order checked by reading, not run on hardware
//
#include <hip/hip_runtime.h>

typedef __attribute__((ext_vector_type(16))) _Float16 v16h;
typedef __attribute__((ext_vector_type(8)))  _Float16 h8;
typedef __attribute__((ext_vector_type(4)))  _Float16 h4;
typedef __attribute__((ext_vector_type(8)))  float    v8f;
typedef __attribute__((ext_vector_type(4)))  float    f32x4;

#define B_DIM 16
#define S_DIM 2048
#define E_DIM 1024

// ---------------------------------------------------------------------------
// Helpers
// ---------------------------------------------------------------------------
__device__ __forceinline__ v8f wmma_f16(v16h a, v16h b, v8f c) {
  return __builtin_amdgcn_wmma_f32_16x16x32_f16(false, a, false, b, (short)0, c,
                                                false, false);
}

// Load 16 f16 in the A/B per-lane K-chunk pattern:
//   halves 0..7  = base[khalf*8 + 0..7]
//   halves 8..15 = base[16 + khalf*8 + 0..7]
__device__ __forceinline__ v16h load_kchunk(const _Float16* __restrict__ base,
                                            int khalf) {
  v16h r;
  *(h8*)&r       = *(const h8*)(base + khalf * 8);
  *((h8*)&r + 1) = *(const h8*)(base + 16 + khalf * 8);
  return r;
}

// ---------------------------------------------------------------------------
// Kernel 0: zero the output accumulator
// ---------------------------------------------------------------------------
__global__ void zero_kernel(float* __restrict__ p, int n) {
  int i = blockIdx.x * blockDim.x + threadIdx.x;
  if (i < n) p[i] = 0.0f;
}

// ---------------------------------------------------------------------------
// Kernel 0b: f32 -> f16 conversion (vectorized x4)
// ---------------------------------------------------------------------------
__global__ void cvt_kernel(const float* __restrict__ src,
                           _Float16* __restrict__ dst, int n4) {
  int i = blockIdx.x * blockDim.x + threadIdx.x;
  if (i < n4) {
    f32x4 v = ((const f32x4*)src)[i];
    h4 h;
    h.x = (_Float16)v.x;
    h.y = (_Float16)v.y;
    h.z = (_Float16)v.z;
    h.w = (_Float16)v.w;
    ((h4*)dst)[i] = h;
  }
}

// ---------------------------------------------------------------------------
// Kernel 1: fused QKV projection.
//   Q = x @ Wq^T + bq   (row-major [B*S, E] f16)
//   K = x @ Wk^T + bk   (row-major [B*S, E] f16)
//   V = x @ Wv^T + bv   (stored TRANSPOSED per batch: Vt[b][e][s] f16)
// One wave computes a 32(M) x 32(N) slab for all three outputs:
// 12 WMMAs per k-step sharing 2 A-tile loads and 6 B-tile loads.
// ---------------------------------------------------------------------------
__global__ __launch_bounds__(256) void qkv_kernel(
    const _Float16* __restrict__ Xh, const _Float16* __restrict__ Wq,
    const _Float16* __restrict__ Wk, const _Float16* __restrict__ Wv,
    const float* __restrict__ bq, const float* __restrict__ bk,
    const float* __restrict__ bv, _Float16* __restrict__ Qh,
    _Float16* __restrict__ Kh, _Float16* __restrict__ Vt) {
  const int wave = threadIdx.x >> 5;
  const int lane = threadIdx.x & 31;
  const int n15 = lane & 15;
  const int khalf = lane >> 4;

  // 1024 M-blocks (32 tokens) x 32 N-blocks (32 features) = 32768 wave-tiles
  const int tile = blockIdx.x * 8 + wave;
  const int mb = tile >> 5;                     // [0, 1024)
  const int nt = (tile & 31) * 2;               // n-tile pair base

  const _Float16* xrow0 = Xh + (size_t)(mb * 32 + n15) * E_DIM;
  const _Float16* xrow1 = xrow0 + (size_t)16 * E_DIM;

  v8f accQ[2][2] = {}, accK[2][2] = {}, accV[2][2] = {};  // [mi][j]

  for (int e0 = 0; e0 < E_DIM; e0 += 32) {
    v16h a0 = load_kchunk(xrow0 + e0, khalf);
    v16h a1 = load_kchunk(xrow1 + e0, khalf);
#pragma unroll
    for (int j = 0; j < 2; ++j) {
      const int f = (nt + j) * 16 + n15;        // output feature = B column
      v16h bqv = load_kchunk(Wq + (size_t)f * E_DIM + e0, khalf);
      v16h bkv = load_kchunk(Wk + (size_t)f * E_DIM + e0, khalf);
      v16h bvv = load_kchunk(Wv + (size_t)f * E_DIM + e0, khalf);
      accQ[0][j] = wmma_f16(a0, bqv, accQ[0][j]);
      accQ[1][j] = wmma_f16(a1, bqv, accQ[1][j]);
      accK[0][j] = wmma_f16(a0, bkv, accK[0][j]);
      accK[1][j] = wmma_f16(a1, bkv, accK[1][j]);
      accV[0][j] = wmma_f16(a0, bvv, accV[0][j]);
      accV[1][j] = wmma_f16(a1, bvv, accV[1][j]);
    }
  }

  // Epilogue: C/D layout -> lane holds column n15, rows r + 8*khalf.
#pragma unroll
  for (int j = 0; j < 2; ++j) {
    const int f = (nt + j) * 16 + n15;
    const float vbq = bq[f], vbk = bk[f], vbv = bv[f];
#pragma unroll
    for (int mi = 0; mi < 2; ++mi) {
#pragma unroll
      for (int r = 0; r < 8; ++r) {
        const int tok = mb * 32 + mi * 16 + 8 * khalf + r;
        Qh[(size_t)tok * E_DIM + f] = (_Float16)(accQ[mi][j][r] + vbq);
        Kh[(size_t)tok * E_DIM + f] = (_Float16)(accK[mi][j][r] + vbk);
        const int bb = tok >> 11;               // tok / S_DIM
        const int ss = tok & (S_DIM - 1);
        Vt[((size_t)bb * E_DIM + f) * S_DIM + ss] =
            (_Float16)(accV[mi][j][r] + vbv);
      }
    }
  }
}

// ---------------------------------------------------------------------------
// Kernel 2: attention for one (batch, 64-query block) per workgroup.
//   Phase 1: P = softmax-input scores (Q Kt / 32) -> LDS (64 x 2048 f16)
//   Phase 2: stable softmax in place, keep exp() unnormalized + 1/rowsum
//   Phase 3: context = P @ V via WMMA; fuse mean-over-queries into
//            atomicAdd on out[b][e].
// ---------------------------------------------------------------------------
__global__ __launch_bounds__(256) void attn_kernel(
    const _Float16* __restrict__ Qh, const _Float16* __restrict__ Kh,
    const _Float16* __restrict__ Vt, float* __restrict__ out) {
  __shared__ _Float16 P[64][S_DIM];             // 256 KB of the 320 KB LDS
  __shared__ float redbuf[64][4];
  __shared__ float rowm[64];
  __shared__ float invl[64];

  const int wave = threadIdx.x >> 5;
  const int lane = threadIdx.x & 31;
  const int n15 = lane & 15;
  const int khalf = lane >> 4;

  const int b = blockIdx.x >> 5;                // 32 query-blocks per batch
  const int q0 = (blockIdx.x & 31) * 64;

  // ---------------- Phase 1: scores ----------------
  // 4 q-subtiles x 16 key-groups (8 key-tiles each), wave-strided.
  // Each wave holds 8 accumulators so the Q (A) tile is reused 8x; the 8
  // K rows sit at immediate offsets (j * 16*E_DIM halves = 32 KB) off one base.
  for (int t = wave; t < 64; t += 8) {
    const int qsub = t >> 4;
    const int kg = t & 15;                      // key-group: 8 tiles = 128 keys
    const _Float16* qrow =
        Qh + (size_t)(b * S_DIM + q0 + qsub * 16 + n15) * E_DIM;
    const _Float16* kbase =
        Kh + (size_t)(b * S_DIM + kg * 128 + n15) * E_DIM;
    v8f acc[8] = {};
    for (int e0 = 0; e0 < E_DIM; e0 += 32) {
      v16h a = load_kchunk(qrow + e0, khalf);
#pragma unroll
      for (int j = 0; j < 8; ++j) {
        v16h kb = load_kchunk(kbase + (size_t)j * 16 * E_DIM + e0, khalf);
        acc[j] = wmma_f16(a, kb, acc[j]);
      }
    }
#pragma unroll
    for (int j = 0; j < 8; ++j) {
#pragma unroll
      for (int r = 0; r < 8; ++r) {
        P[qsub * 16 + 8 * khalf + r][(kg * 8 + j) * 16 + n15] =
            (_Float16)(acc[j][r] * 0.03125f);   // 1/sqrt(1024)
      }
    }
  }
  __syncthreads();

  // ---------------- Phase 2: softmax ----------------
  {
    const int row = threadIdx.x >> 2;           // 64 rows
    const int part = threadIdx.x & 3;           // 4 x 512 columns
    const int base = part * 512;

    float mx = -1e30f;
    for (int i = 0; i < 512; i += 8) {
      h8 c = *(const h8*)&P[row][base + i];
#pragma unroll
      for (int j = 0; j < 8; ++j) mx = fmaxf(mx, (float)c[j]);
    }
    redbuf[row][part] = mx;
    __syncthreads();
    if (part == 0) {
      rowm[row] = fmaxf(fmaxf(redbuf[row][0], redbuf[row][1]),
                        fmaxf(redbuf[row][2], redbuf[row][3]));
    }
    __syncthreads();

    const float m = rowm[row];
    float sum = 0.0f;
    for (int i = 0; i < 512; i += 8) {
      h8 c = *(const h8*)&P[row][base + i];
      h8 e;
#pragma unroll
      for (int j = 0; j < 8; ++j) {
        float ev = __expf((float)c[j] - m);
        sum += ev;
        e[j] = (_Float16)ev;
      }
      *(h8*)&P[row][base + i] = e;
    }
    redbuf[row][part] = sum;
    __syncthreads();
    if (part == 0) {
      invl[row] = 1.0f / (redbuf[row][0] + redbuf[row][1] + redbuf[row][2] +
                          redbuf[row][3]);
    }
    __syncthreads();
  }

  // ---------------- Phase 3: context + fused mean ----------------
  // Each wave covers 64 output columns per pass; 8 waves x 2 passes = 1024.
  // 16 accumulator tiles: 4 P-tiles (LDS) x 4 V-tiles (global) per k-step.
  for (int pass = 0; pass < 2; ++pass) {
    const int colbase = pass * 512 + wave * 64;
    const _Float16* vbase = Vt + ((size_t)b * E_DIM + colbase + n15) * S_DIM;
    v8f acc[4][4] = {};                         // [qsub][csub]

    for (int k0 = 0; k0 < S_DIM; k0 += 32) {
      v16h aT[4];
#pragma unroll
      for (int qsub = 0; qsub < 4; ++qsub) {
        // A operand from LDS (probabilities), row = q, K = keys
        aT[qsub] = load_kchunk(&P[qsub * 16 + n15][k0], khalf);
      }
#pragma unroll
      for (int c = 0; c < 4; ++c) {
        // V-tile columns at immediate offsets (c * 16*S_DIM halves = 64 KB)
        v16h bv = load_kchunk(vbase + (size_t)c * 16 * S_DIM + k0, khalf);
#pragma unroll
        for (int qsub = 0; qsub < 4; ++qsub)
          acc[qsub][c] = wmma_f16(aT[qsub], bv, acc[qsub][c]);
      }
    }

    // Reduce over the 64 queries of this block, normalize by row-sum and
    // fold in the 1/S of the sequence mean; accumulate into out[b][col].
#pragma unroll
    for (int c = 0; c < 4; ++c) {
      const int col = colbase + c * 16 + n15;
      float partial = 0.0f;
#pragma unroll
      for (int qsub = 0; qsub < 4; ++qsub) {
#pragma unroll
        for (int r = 0; r < 8; ++r) {
          const int qrow = qsub * 16 + 8 * khalf + r;
          partial += acc[qsub][c][r] * invl[qrow];
        }
      }
      partial += __shfl_xor(partial, 16, 32);   // combine the two half-waves
      if (lane < 16)
        atomicAdd(&out[b * E_DIM + col], partial * (1.0f / (float)S_DIM));
    }
  }
}

// ---------------------------------------------------------------------------
// Host-side launcher
// ---------------------------------------------------------------------------
extern "C" void kernel_launch(void* const* d_in, const int* in_sizes, int n_in,
                              void* d_out, int out_size, void* d_ws,
                              size_t ws_size, hipStream_t stream) {
  const float* x = (const float*)d_in[0];
  const float* wq = (const float*)d_in[1];
  const float* bq = (const float*)d_in[2];
  const float* wk = (const float*)d_in[3];
  const float* bk = (const float*)d_in[4];
  const float* wv = (const float*)d_in[5];
  const float* bv = (const float*)d_in[6];
  float* out = (float*)d_out;

  const size_t BSE = (size_t)B_DIM * S_DIM * E_DIM;  // 33,554,432
  const size_t EE = (size_t)E_DIM * E_DIM;           // 1,048,576

  _Float16* Xh = (_Float16*)d_ws;
  _Float16* Wqh = Xh + BSE;
  _Float16* Wkh = Wqh + EE;
  _Float16* Wvh = Wkh + EE;
  _Float16* Qh = Wvh + EE;
  _Float16* Kh = Qh + BSE;
  _Float16* Vt = Kh + BSE;  // total ~262 MiB of workspace

  // 0) zero output accumulator (atomics accumulate into it)
  zero_kernel<<<(B_DIM * E_DIM + 255) / 256, 256, 0, stream>>>(
      out, B_DIM * E_DIM);

  // 1) f32 -> f16 conversions
  cvt_kernel<<<(int)((BSE / 4 + 255) / 256), 256, 0, stream>>>(x, Xh,
                                                               (int)(BSE / 4));
  cvt_kernel<<<(int)((EE / 4 + 255) / 256), 256, 0, stream>>>(wq, Wqh,
                                                              (int)(EE / 4));
  cvt_kernel<<<(int)((EE / 4 + 255) / 256), 256, 0, stream>>>(wk, Wkh,
                                                              (int)(EE / 4));
  cvt_kernel<<<(int)((EE / 4 + 255) / 256), 256, 0, stream>>>(wv, Wvh,
                                                              (int)(EE / 4));

  // 2) fused QKV projection: 1024 M-blocks x 32 N-blocks = 32768 wave-tiles
  qkv_kernel<<<4096, 256, 0, stream>>>(Xh, Wqh, Wkh, Wvh, bq, bk, bv, Qh, Kh,
                                       Vt);

  // 3) attention + fused sequence mean: 16 batches x 32 query-blocks
  attn_kernel<<<B_DIM * 32, 256, 0, stream>>>(Qh, Kh, Vt, out);
}
